// Actor_42528766165189
// MI455X (gfx1250) — compile-verified, matrix operand order checked by reading, not run on hardware
//
#include <hip/hip_runtime.h>
#include <hip/hip_bf16.h>

typedef __attribute__((ext_vector_type(16))) __bf16 v16bf;
typedef __attribute__((ext_vector_type(8)))  __bf16 v8bf;
typedef __attribute__((ext_vector_type(4)))  __bf16 v4bf;
typedef __attribute__((ext_vector_type(8)))  float  v8f;

#define B_TOT  1024
#define T_DIM  256
#define D_DIM  64
#define H_DIM  512
#define BB_DIM 128
#define A_DIM  8

// LDS row strides (bf16 elements), padded, rows stay 16B-aligned
#define ZS 592   // 576 + 16
#define GS 144   // 128 + 16
#define YS 528   // 512 + 16

// Swizzled bf16 weight layout in d_ws (element offsets)
#define SZ_WBB (BB_DIM * (D_DIM + H_DIM))   // 73728
#define SZ_FF  (H_DIM * BB_DIM)             // 65536
#define SZ_F1  (H_DIM * H_DIM)              // 262144
#define OFF_WBB 0
#define OFF_FF1 (OFF_WBB + SZ_WBB)
#define OFF_FF2 (OFF_FF1 + SZ_FF)
#define OFF_TA  (OFF_FF2 + SZ_FF)
#define OFF_TB  (OFF_TA  + SZ_FF)
#define OFF_F1  (OFF_TB  + SZ_FF)

// ---------------------------------------------------------------------------
// Pre-swizzle: W is [N,K] row-major f32. Build B = W^T tiles (K=32 x N=16) in
// the CDNA5 16-bit B-fragment lane order: lane l(0..15) -> col n=l,
// K {0..7,16..23}; lane l(16..31) -> col n=l-16, K {8..15,24..31}.
// ---------------------------------------------------------------------------
__global__ void swz_weights(const float* __restrict__ W, __bf16* __restrict__ dst,
                            int N, int K) {
  int idx = blockIdx.x * blockDim.x + threadIdx.x;
  int total = N * K;
  if (idx >= total) return;
  int e    = idx & 15;
  int l    = (idx >> 4) & 31;
  int tile = idx >> 9;
  int ktiles = K >> 5;
  int kt = tile % ktiles;
  int nt = tile / ktiles;
  int n  = nt * 16 + (l & 15);
  int k  = kt * 32 + ((l >> 4) << 3) + (e < 8 ? e : e + 8);
  dst[idx] = (__bf16)W[n * K + k];
}

// ---------------------------------------------------------------------------
__device__ __forceinline__ v16bf ldg_bfrag(const __bf16* p) {
  return *reinterpret_cast<const v16bf*>(p);   // 32B/lane, coalesced global_load
}

// A-fragment (16x32 bf16) from LDS row-major buffer: two ds_load_b128/lane.
__device__ __forceinline__ v16bf lds_afrag(const __bf16* buf, int lane, int k0, int stride) {
  int m  = lane & 15;
  int ko = k0 + ((lane >> 4) << 3);
  const __bf16* p = buf + m * stride + ko;
  v8bf lo = *reinterpret_cast<const v8bf*>(p);
  v8bf hi = *reinterpret_cast<const v8bf*>(p + 16);
  return __builtin_shufflevector(lo, hi, 0,1,2,3,4,5,6,7,8,9,10,11,12,13,14,15);
}

__device__ __forceinline__ v8f wmma_bf16(v16bf a, v16bf b, v8f c) {
  return __builtin_amdgcn_wmma_f32_16x16x32_bf16(false, a, false, b, (short)0, c,
                                                 false, false);
}

// Branch-free transcendentals (v_exp_f32 + v_rcp_f32, saturate correctly):
__device__ __forceinline__ float fast_tanh(float x) {
  float e = __expf(2.0f * x);                       // inf for large x -> tanh=1
  return 1.0f - 2.0f * __builtin_amdgcn_rcpf(e + 1.0f);
}
__device__ __forceinline__ float fast_sigmoid(float x) {
  return __builtin_amdgcn_rcpf(1.0f + __expf(-x));
}

// ---------------------------------------------------------------------------
// Main CfC kernel: one block = 16 batch rows, 8 waves, full T-step scan.
// ---------------------------------------------------------------------------
__global__ __launch_bounds__(256, 1)
void cfc_scan_kernel(const float* __restrict__ state,
                     const __bf16* __restrict__ wsz,
                     const float* __restrict__ bbb,
                     const float* __restrict__ bff1,
                     const float* __restrict__ bff2,
                     const float* __restrict__ bta,
                     const float* __restrict__ btb,
                     const float* __restrict__ bf1,
                     const float* __restrict__ Wf2,
                     const float* __restrict__ bf2,
                     float* __restrict__ out)
{
  __shared__ __align__(16) __bf16 zbuf[16 * ZS];   // [x_t | h]
  __shared__ __align__(16) __bf16 gbuf[16 * GS];   // backbone activations
  __shared__ __align__(16) __bf16 ybuf[16 * YS];   // fc1 output

  const int tid  = threadIdx.x;
  const int lane = tid & 31;
  const int wave = tid >> 5;
  const int b0   = blockIdx.x * 16;

  // h0 = 0
  for (int i = tid; i < 16 * H_DIM; i += 256) {
    int r = i >> 9, c = i & 511;
    zbuf[r * ZS + D_DIM + c] = (__bf16)0.0f;
  }

  const int n_bb  = wave * 16 + (lane & 15);   // backbone output column
  const int mbase = (lane >> 4) << 3;          // C/D row base for this lane

  // Hoist loop-invariant biases into registers.
  const float bias_bb = bbb[n_bb];
  float pb1[4], pb2[4], pbab[4], pbf1[4];
#pragma unroll
  for (int j = 0; j < 4; ++j) {
    int n = (wave + 8 * j) * 16 + (lane & 15);
    pb1[j]  = bff1[n];
    pb2[j]  = bff2[n];
    pbab[j] = bta[n] + btb[n];
    pbf1[j] = bf1[n];
  }

  // x_t staging: thread -> (row r, 4 cols); one float4 + packed bf16x4 store.
  const int xr  = tid >> 4;
  const int xc4 = (tid & 15) << 2;
  const float* xrow = state + ((size_t)(b0 + xr) * T_DIM) * D_DIM + xc4;

  for (int t = 0; t < T_DIM; ++t) {
    // Opaque zero offset per iteration: defeats LICM of the (t-invariant)
    // weight-fragment loads WITHOUT destroying the pointer's address-space
    // provenance, so they stay global_load_b128 (not flat_) and are re-read
    // from the shared, L2-resident swizzled arrays instead of spilling.
    int toff = 0;
    asm volatile("" : "+s"(toff));
    const __bf16* Wbb_s = wsz + OFF_WBB + toff;
    const __bf16* Wp1_s = wsz + OFF_FF1 + toff;
    const __bf16* Wp2_s = wsz + OFF_FF2 + toff;
    const __bf16* Wta_s = wsz + OFF_TA  + toff;
    const __bf16* Wtb_s = wsz + OFF_TB  + toff;

    // ---- stage x_t into z[:, 0:64) as bf16 ----
    {
      float4 xv = *reinterpret_cast<const float4*>(xrow + (size_t)t * D_DIM);
      v4bf o;
      o[0] = (__bf16)xv.x; o[1] = (__bf16)xv.y;
      o[2] = (__bf16)xv.z; o[3] = (__bf16)xv.w;
      *reinterpret_cast<v4bf*>(&zbuf[xr * ZS + xc4]) = o;
      if (t + 1 < T_DIM)
        __builtin_prefetch(xrow + (size_t)(t + 1) * D_DIM, 0, 3);  // global_prefetch_b8
    }
    __syncthreads();   // x_t + h(t-1) visible

    // ---- backbone: g = lecun_tanh(z @ Wbb^T + bbb); wave owns N-tile = wave
    // software-pipelined one stage ahead so loads overlap WMMA.
    {
      v8f acc = {};
      const __bf16* wp = Wbb_s + (size_t)(wave * 18) * 512 + lane * 16;
      v16bf a_cur = lds_afrag(zbuf, lane, 0, ZS);
      v16bf b_cur = ldg_bfrag(wp);
#pragma unroll
      for (int kt = 0; kt < 17; ++kt) {
        v16bf a_nxt = lds_afrag(zbuf, lane, (kt + 1) * 32, ZS);
        v16bf b_nxt = ldg_bfrag(wp + 512);
        acc = wmma_bf16(a_cur, b_cur, acc);
        a_cur = a_nxt; b_cur = b_nxt; wp += 512;
      }
      acc = wmma_bf16(a_cur, b_cur, acc);
#pragma unroll
      for (int v = 0; v < 8; ++v) {
        float g = 1.7159f * fast_tanh(0.666f * (acc[v] + bias_bb));
        gbuf[(mbase + v) * GS + n_bb] = (__bf16)g;
      }
    }
    __syncthreads();   // g visible

    // ---- four projections + CfC gate; wave owns H-tiles {wave, +8, +16, +24}
    // A-fragments of g are invariant across the 4 tiles: load once.
    v16bf ag[4];
#pragma unroll
    for (int kt = 0; kt < 4; ++kt) ag[kt] = lds_afrag(gbuf, lane, kt * 32, GS);

#pragma unroll
    for (int j = 0; j < 4; ++j) {
      const int nt = wave + 8 * j;
      const size_t tb = (size_t)(nt * 4) * 512 + lane * 16;
      v8f aF1 = {}, aF2 = {}, aTA = {}, aTB = {};
      v16bf c1 = ldg_bfrag(Wp1_s + tb), c2 = ldg_bfrag(Wp2_s + tb);
      v16bf ca = ldg_bfrag(Wta_s + tb), cb = ldg_bfrag(Wtb_s + tb);
#pragma unroll
      for (int kt = 0; kt < 4; ++kt) {
        v16bf n1, n2, na, nb;
        if (kt < 3) {
          size_t o = tb + (size_t)(kt + 1) * 512;
          n1 = ldg_bfrag(Wp1_s + o); n2 = ldg_bfrag(Wp2_s + o);
          na = ldg_bfrag(Wta_s + o); nb = ldg_bfrag(Wtb_s + o);
        }
        aF1 = wmma_bf16(ag[kt], c1, aF1);
        aF2 = wmma_bf16(ag[kt], c2, aF2);
        aTA = wmma_bf16(ag[kt], ca, aTA);
        aTB = wmma_bf16(ag[kt], cb, aTB);
        c1 = n1; c2 = n2; ca = na; cb = nb;
      }
      const int n = nt * 16 + (lane & 15);
#pragma unroll
      for (int v = 0; v < 8; ++v) {
        float f1 = fast_tanh(aF1[v] + pb1[j]);
        float f2 = fast_tanh(aF2[v] + pb2[j]);
        float ti = fast_sigmoid(aTA[v] + aTB[v] + pbab[j]);   // ts = 1.0
        float h  = f1 + ti * (f2 - f1);
        zbuf[(mbase + v) * ZS + D_DIM + n] = (__bf16)h;
      }
    }
    // next-iteration barrier orders h-writes vs z-reads
  }
  __syncthreads();   // final h visible

  // ---- fc1: y = tanh(h @ Wf1^T + bf1) via WMMA, pipelined ----
#pragma unroll
  for (int j = 0; j < 4; ++j) {
    const int nt = wave + 8 * j;
    v8f acc = {};
    const __bf16* wp = wsz + OFF_F1 + (size_t)(nt * 16) * 512 + lane * 16;
    v16bf a_cur = lds_afrag(zbuf, lane, D_DIM, ZS);
    v16bf b_cur = ldg_bfrag(wp);
#pragma unroll
    for (int kt = 0; kt < 15; ++kt) {
      v16bf a_nxt = lds_afrag(zbuf, lane, D_DIM + (kt + 1) * 32, ZS);
      v16bf b_nxt = ldg_bfrag(wp + 512);
      acc = wmma_bf16(a_cur, b_cur, acc);
      a_cur = a_nxt; b_cur = b_nxt; wp += 512;
    }
    acc = wmma_bf16(a_cur, b_cur, acc);
    const int n = nt * 16 + (lane & 15);
#pragma unroll
    for (int v = 0; v < 8; ++v)
      ybuf[(mbase + v) * YS + n] = (__bf16)fast_tanh(acc[v] + pbf1[j]);
  }
  __syncthreads();

  // ---- fc2: out = tanh(y @ Wf2^T + bf2), N=8 -> vectorized dot products ----
  if (tid < 16 * A_DIM) {
    const int r = tid >> 3, c = tid & 7;
    float s = bf2[c];
    const __bf16* yr = ybuf + r * YS;
    const float* w = Wf2 + c * H_DIM;
    for (int k = 0; k < H_DIM; k += 8) {
      v8bf yv = *reinterpret_cast<const v8bf*>(yr + k);
#pragma unroll
      for (int u = 0; u < 8; ++u)
        s = fmaf((float)yv[u], w[k + u], s);
    }
    out[(size_t)(b0 + r) * A_DIM + c] = fast_tanh(s);
  }
}

// ---------------------------------------------------------------------------
extern "C" void kernel_launch(void* const* d_in, const int* in_sizes, int n_in,
                              void* d_out, int out_size, void* d_ws, size_t ws_size,
                              hipStream_t stream) {
  const float* state = (const float*)d_in[0];
  const float* Wbb   = (const float*)d_in[1];
  const float* bbb   = (const float*)d_in[2];
  const float* Wff1  = (const float*)d_in[3];
  const float* bff1  = (const float*)d_in[4];
  const float* Wff2  = (const float*)d_in[5];
  const float* bff2  = (const float*)d_in[6];
  const float* Wta   = (const float*)d_in[7];
  const float* bta   = (const float*)d_in[8];
  const float* Wtb   = (const float*)d_in[9];
  const float* btb   = (const float*)d_in[10];
  const float* Wf1   = (const float*)d_in[11];
  const float* bf1   = (const float*)d_in[12];
  const float* Wf2   = (const float*)d_in[13];
  const float* bf2   = (const float*)d_in[14];

  __bf16* ws = (__bf16*)d_ws;

  swz_weights<<<(SZ_WBB + 255) / 256, 256, 0, stream>>>(Wbb,  ws + OFF_WBB, BB_DIM, D_DIM + H_DIM);
  swz_weights<<<(SZ_FF  + 255) / 256, 256, 0, stream>>>(Wff1, ws + OFF_FF1, H_DIM, BB_DIM);
  swz_weights<<<(SZ_FF  + 255) / 256, 256, 0, stream>>>(Wff2, ws + OFF_FF2, H_DIM, BB_DIM);
  swz_weights<<<(SZ_FF  + 255) / 256, 256, 0, stream>>>(Wta,  ws + OFF_TA,  H_DIM, BB_DIM);
  swz_weights<<<(SZ_FF  + 255) / 256, 256, 0, stream>>>(Wtb,  ws + OFF_TB,  H_DIM, BB_DIM);
  swz_weights<<<(SZ_F1  + 255) / 256, 256, 0, stream>>>(Wf1,  ws + OFF_F1,  H_DIM, H_DIM);

  cfc_scan_kernel<<<B_TOT / 16, 256, 0, stream>>>(
      state, ws, bbb, bff1, bff2, bta, btb, bf1, Wf2, bf2, (float*)d_out);
}
